// ChimeraAutonomousSwarm_9234179686578
// MI455X (gfx1250) — matrix operand order repaired and checked by reading, not compiled, not run on hardware
//
#include <hip/hip_runtime.h>
#include <hip/hip_bf16.h>

// ---------------------------------------------------------------------------
// Problem sizes (from reference)
// ---------------------------------------------------------------------------
#define T_TOK 4096
#define DMODEL 1024
#define DFF 2048
#define NEXP 8
#define BT 32                      // token tile per block

typedef __bf16 bf16_t;
typedef __attribute__((ext_vector_type(16))) __bf16 v16bf;
typedef __attribute__((ext_vector_type(8)))  __bf16 v8bf;
typedef __attribute__((ext_vector_type(8)))  float  v8f;
typedef __attribute__((ext_vector_type(4)))  int    v4i;

// LDS tile strides (bf16 elements), padded to stagger banks
#define XSTR 1032                  // x tile row stride (D=1024 + 8)
#define HSTR 2056                  // h tile row stride (DFF=2048 + 8)
#define XS_BYTES (BT * XSTR * 2)   // 66048
#define HS_BYTES (BT * HSTR * 2)   // 131584
#define SS_BYTES (BT * 4)
#define SMEM_BYTES (XS_BYTES + HS_BYTES + SS_BYTES)   // ~197.8 KB < 320 KB

// gfx1250 async global->LDS DMA, if the toolchain exposes it
#if defined(__has_builtin)
# if __has_builtin(__builtin_amdgcn_global_load_async_to_lds_b128)
#  define USE_ASYNC_LDS 1
# endif
# if __has_builtin(__builtin_amdgcn_s_wait_asynccnt)
#  define HAVE_WAIT_ASYNC 1
# endif
#endif

// ---------------------------------------------------------------------------
// WMMA fragment loaders (CDNA5 16x16x32 bf16 layouts, ISA 7.12.2)
// A 16x32: lane L holds row M=L%16; K segs [8*half..+7] and [16+8*half..+7]
// B 32x16: lane L holds col N=L%16; K segs [8*half..+7] and [16+8*half..+7]
// ---------------------------------------------------------------------------
__device__ __forceinline__ v16bf load_frag_a(const bf16_t* __restrict__ base,
                                             int stride, int row0, int k, int lane) {
  const int mm = lane & 15;
  const int hb = lane >> 4;
  const bf16_t* p = base + (row0 + mm) * stride + k + 8 * hb;
  v8bf lo = *(const v8bf*)(p);
  v8bf hi = *(const v8bf*)(p + 16);
  v16bf out;
#pragma unroll
  for (int i = 0; i < 8; ++i) { out[i] = lo[i]; out[8 + i] = hi[i]; }
  return out;
}

__device__ __forceinline__ v16bf load_frag_b(const bf16_t* __restrict__ W,
                                             int ldk, int n0, int k, int lane) {
  const int nn = lane & 15;
  const int hb = lane >> 4;
  const bf16_t* p = W + (size_t)(n0 + nn) * ldk + k + 8 * hb;
  v8bf lo = *(const v8bf*)(p);
  v8bf hi = *(const v8bf*)(p + 16);
  v16bf out;
#pragma unroll
  for (int i = 0; i < 8; ++i) { out[i] = lo[i]; out[8 + i] = hi[i]; }
  return out;
}

__device__ __forceinline__ v8f wmma_bf16(v16bf a, v16bf b, v8f c) {
  return __builtin_amdgcn_wmma_f32_16x16x32_bf16(
      /*neg_a=*/false, a, /*neg_b=*/false, b,
      /*c_mod=*/(short)0, c, /*reuse_a=*/false, /*reuse_b=*/false);
}

// fast sigmoid: v_exp_f32 + v_rcp_f32 (avoid IEEE division chains)
__device__ __forceinline__ float fast_sigmoid(float v) {
  return __builtin_amdgcn_rcpf(1.f + __expf(-v));
}

// ---------------------------------------------------------------------------
// Kernel 1: f32 -> bf16 bulk convert (weights + x), vectorized float4
// ---------------------------------------------------------------------------
__global__ void f32_to_bf16_kernel(const float* __restrict__ src,
                                   bf16_t* __restrict__ dst, size_t n4) {
  size_t i = (size_t)blockIdx.x * blockDim.x + threadIdx.x;
  size_t stride = (size_t)gridDim.x * blockDim.x;
  for (; i < n4; i += stride) {
    float4 v = ((const float4*)src)[i];
    bf16_t* o = dst + i * 4;
    o[0] = (bf16_t)v.x; o[1] = (bf16_t)v.y; o[2] = (bf16_t)v.z; o[3] = (bf16_t)v.w;
  }
}

// ---------------------------------------------------------------------------
// Kernel 2: probe -> scale[T,E].  One wave per token (wave32).
// tau = tau_base + gamma * silu(w_depth * 0.5); scale = sigmoid(l)*(l>tau)
// ---------------------------------------------------------------------------
__global__ __launch_bounds__(256) void probe_kernel(
    const float* __restrict__ x, const float* __restrict__ Wp,
    const float* __restrict__ bp, const float* __restrict__ taub,
    const float* __restrict__ gam, const float* __restrict__ wdep,
    float* __restrict__ scale) {
  const int wave = (int)((blockIdx.x * blockDim.x + threadIdx.x) >> 5);
  const int lane = threadIdx.x & 31;
  if (wave >= T_TOK) return;
  const float* xr = x + (size_t)wave * DMODEL;
  float acc[NEXP];
#pragma unroll
  for (int e = 0; e < NEXP; ++e) acc[e] = 0.f;
  for (int i = lane * 4; i < DMODEL; i += 32 * 4) {
    float4 xv = *(const float4*)(xr + i);
#pragma unroll
    for (int e = 0; e < NEXP; ++e) {
      float4 wv = *(const float4*)(Wp + e * DMODEL + i);
      acc[e] += xv.x * wv.x + xv.y * wv.y + xv.z * wv.z + xv.w * wv.w;
    }
  }
#pragma unroll
  for (int e = 0; e < NEXP; ++e)
#pragma unroll
    for (int off = 16; off; off >>= 1)
      acc[e] += __shfl_xor(acc[e], off, 32);
  if (lane == 0) {
    const float dr = wdep[0] * 0.5f;
    const float tau = taub[0] + gam[0] * (dr * fast_sigmoid(dr));
#pragma unroll
    for (int e = 0; e < NEXP; ++e) {
      float lg = acc[e] + bp[e];
      scale[(size_t)wave * NEXP + e] = (lg > tau) ? fast_sigmoid(lg) : 0.f;
    }
  }
}

// ---------------------------------------------------------------------------
// Kernel 3: fused expert SwiGLU MLP.
// grid = NEXP * (T/BT) blocks, 256 threads = 8 waves.
// Per block: async-DMA x tile (32x1024 bf16) into LDS, compute
// h = up*silu(gate) (32x2048 bf16) in LDS via WMMA, then down-project and
// atomically scale-combine into out.
// Each wave owns BOTH 16-row M halves of its N tile, so every global B
// fragment feeds two WMMAs (halves L2 weight traffic per matrix op).
// ---------------------------------------------------------------------------
__global__ __launch_bounds__(256) void moe_fused_kernel(
    const bf16_t* __restrict__ xbf, const bf16_t* __restrict__ wup,
    const bf16_t* __restrict__ wgt, const bf16_t* __restrict__ wdn,
    const float* __restrict__ scale, float* __restrict__ out) {
  extern __shared__ char smem[];
  bf16_t* xs = (bf16_t*)smem;                        // [BT][XSTR]
  bf16_t* hs = (bf16_t*)(smem + XS_BYTES);           // [BT][HSTR]
  float*  ss = (float*)(smem + XS_BYTES + HS_BYTES); // [BT]

  const int ntt = T_TOK / BT;          // 128 token tiles
  const int e   = blockIdx.x / ntt;
  const int tt  = blockIdx.x % ntt;
  const int t0  = tt * BT;

  const bf16_t* Wu = wup + (size_t)e * DFF * DMODEL;
  const bf16_t* Wg = wgt + (size_t)e * DFF * DMODEL;
  const bf16_t* Wd = wdn + (size_t)e * DMODEL * DFF;

  const int tid = threadIdx.x;

  // stage x tile into LDS
#if defined(USE_ASYNC_LDS)
  // direct global->LDS DMA (GLOBAL_LOAD_ASYNC_TO_LDS_B128), ASYNCcnt-tracked
  for (int idx = tid; idx < BT * (DMODEL / 8); idx += 256) {
    int row = idx / (DMODEL / 8);
    int c8  = idx % (DMODEL / 8);
    const bf16_t* g = xbf + (size_t)(t0 + row) * DMODEL + c8 * 8;
    bf16_t* l = xs + row * XSTR + c8 * 8;
    __builtin_amdgcn_global_load_async_to_lds_b128(
        (__attribute__((address_space(1))) v4i*)g,
        (__attribute__((address_space(3))) v4i*)l, 0, 0);
  }
# if defined(HAVE_WAIT_ASYNC)
  __builtin_amdgcn_s_wait_asynccnt(0);
# endif
#else
  for (int idx = tid; idx < BT * (DMODEL / 8); idx += 256) {
    int row = idx / (DMODEL / 8);
    int c8  = idx % (DMODEL / 8);
    v8bf v = *(const v8bf*)(xbf + (size_t)(t0 + row) * DMODEL + c8 * 8);
    *(v8bf*)(xs + row * XSTR + c8 * 8) = v;
  }
#endif
  if (tid < BT) ss[tid] = scale[(size_t)(t0 + tid) * NEXP + e];
  __syncthreads();

  const int wave = tid >> 5;
  const int lane = tid & 31;
  const int nn = lane & 15;
  const int hb = lane >> 4;

  // ---- Phase 1: up & gate GEMMs + SwiGLU -> hs ----
  for (int nt = wave; nt < DFF / 16; nt += 8) {
    v8f au0 = {}, ag0 = {}, au1 = {}, ag1 = {};
#pragma unroll 4
    for (int k = 0; k < DMODEL; k += 32) {
      v16bf a0 = load_frag_a(xs, XSTR, 0,  k, lane);
      v16bf a1 = load_frag_a(xs, XSTR, 16, k, lane);
      v16bf bu = load_frag_b(Wu, DMODEL, nt * 16, k, lane);
      v16bf bg = load_frag_b(Wg, DMODEL, nt * 16, k, lane);
      au0 = wmma_bf16(a0, bu, au0);
      au1 = wmma_bf16(a1, bu, au1);
      ag0 = wmma_bf16(a0, bg, ag0);
      ag1 = wmma_bf16(a1, bg, ag1);
    }
#pragma unroll
    for (int r = 0; r < 8; ++r) {
      float g0 = ag0[r];
      float h0 = au0[r] * g0 * __builtin_amdgcn_rcpf(1.f + __expf(-g0));
      hs[(r + 8 * hb) * HSTR + nt * 16 + nn] = (bf16_t)h0;
      float g1 = ag1[r];
      float h1 = au1[r] * g1 * __builtin_amdgcn_rcpf(1.f + __expf(-g1));
      hs[(16 + r + 8 * hb) * HSTR + nt * 16 + nn] = (bf16_t)h1;
    }
  }
  __syncthreads();

  // ---- Phase 2: down GEMM + scale-combine ----
  for (int nt = wave; nt < DMODEL / 16; nt += 8) {
    v8f ad0 = {}, ad1 = {};
#pragma unroll 4
    for (int k = 0; k < DFF; k += 32) {
      v16bf a0 = load_frag_a(hs, HSTR, 0,  k, lane);
      v16bf a1 = load_frag_a(hs, HSTR, 16, k, lane);
      v16bf b  = load_frag_b(Wd, DFF, nt * 16, k, lane);
      ad0 = wmma_bf16(a0, b, ad0);
      ad1 = wmma_bf16(a1, b, ad1);
    }
#pragma unroll
    for (int r = 0; r < 8; ++r) {
      int row0 = r + 8 * hb;
      int row1 = 16 + r + 8 * hb;
      atomicAdd(out + (size_t)(t0 + row0) * DMODEL + nt * 16 + nn, ad0[r] * ss[row0]);
      atomicAdd(out + (size_t)(t0 + row1) * DMODEL + nt * 16 + nn, ad1[r] * ss[row1]);
    }
  }
}

// ---------------------------------------------------------------------------
// Launch
// ---------------------------------------------------------------------------
static inline size_t align256(size_t x) { return (x + 255) & ~(size_t)255; }

extern "C" void kernel_launch(void* const* d_in, const int* in_sizes, int n_in,
                              void* d_out, int out_size, void* d_ws, size_t ws_size,
                              hipStream_t stream) {
  const float* x    = (const float*)d_in[0];
  const float* Wp   = (const float*)d_in[1];
  const float* bp   = (const float*)d_in[2];
  const float* Wu   = (const float*)d_in[3];
  const float* Wg   = (const float*)d_in[4];
  const float* Wd   = (const float*)d_in[5];
  const float* taub = (const float*)d_in[6];
  const float* gam  = (const float*)d_in[7];
  const float* wdep = (const float*)d_in[8];
  float* out = (float*)d_out;

  // workspace carve-up
  char* ws = (char*)d_ws;
  const size_t nX = (size_t)T_TOK * DMODEL;
  const size_t nW = (size_t)NEXP * DFF * DMODEL;     // same for up/gate/down
  bf16_t* xbf = (bf16_t*)ws;                 ws += align256(nX * 2);
  bf16_t* wub = (bf16_t*)ws;                 ws += align256(nW * 2);
  bf16_t* wgb = (bf16_t*)ws;                 ws += align256(nW * 2);
  bf16_t* wdb = (bf16_t*)ws;                 ws += align256(nW * 2);
  float*  scl = (float*)ws;                  ws += align256((size_t)T_TOK * NEXP * 4);
  (void)ws_size; (void)in_sizes; (void)n_in;

  (void)hipMemsetAsync(d_out, 0, (size_t)out_size * sizeof(float), stream);

  // bf16 conversions (weights will then live in L2: 100 MB < 192 MB)
  f32_to_bf16_kernel<<<2048, 256, 0, stream>>>(Wu, wub, nW / 4);
  f32_to_bf16_kernel<<<2048, 256, 0, stream>>>(Wg, wgb, nW / 4);
  f32_to_bf16_kernel<<<2048, 256, 0, stream>>>(Wd, wdb, nW / 4);
  f32_to_bf16_kernel<<<512, 256, 0, stream>>>(x, xbf, nX / 4);

  // probe: one wave32 per token
  probe_kernel<<<(T_TOK * 32) / 256, 256, 0, stream>>>(x, Wp, bp, taub, gam, wdep, scl);

  // fused experts: needs > 64KB dynamic LDS (CDNA5: 320KB/WGP)
  (void)hipFuncSetAttribute((const void*)moe_fused_kernel,
                            hipFuncAttributeMaxDynamicSharedMemorySize, SMEM_BYTES);
  moe_fused_kernel<<<NEXP * (T_TOK / BT), 256, SMEM_BYTES, stream>>>(
      xbf, wub, wgb, wdb, scl, out);
}
